// DeepCrysTet_5677946765689
// MI455X (gfx1250) — matrix operand order repaired and checked by reading, not compiled, not run on hardware
//
#include <hip/hip_runtime.h>
#include <hip/hip_bf16.h>
#include <math.h>

// ---------------------------------------------------------------------------
// DeepCrysTet forward for MI455X (gfx1250, wave32).
// Heavy layers -> bf16 WMMA GEMM (v_wmma_f32_16x16x32_bf16), BN+ReLU fused
// into operand staging (float4 global loads -> packed bf16 b64 LDS stores),
// batch stats fused into the GEMM epilogue with a branch-free full-tile path.
// Tiny front-end layers -> VALU + LDS atomics.
// ---------------------------------------------------------------------------

typedef __attribute__((ext_vector_type(8)))  float  v8f;
typedef __attribute__((ext_vector_type(16))) __bf16 v16bf;

#define EPSBN 1e-5f
#define ENC_NEGINF 0x007FFFFFu

__device__ __forceinline__ unsigned short f2bf(float f) {
  unsigned u = __float_as_uint(f);
  unsigned r = u + 0x7FFFu + ((u >> 16) & 1u);   // round-to-nearest-even
  return (unsigned short)(r >> 16);
}
__device__ __forceinline__ uint2 pack4bf(float a, float b, float c, float d) {
  uint2 r;
  r.x = (unsigned)f2bf(a) | ((unsigned)f2bf(b) << 16);
  r.y = (unsigned)f2bf(c) | ((unsigned)f2bf(d) << 16);
  return r;
}
__device__ __forceinline__ unsigned encf(float f) {
  unsigned u = __float_as_uint(f);
  return (u & 0x80000000u) ? ~u : (u | 0x80000000u);
}
__device__ __forceinline__ float decf(unsigned e) {
  unsigned u = (e & 0x80000000u) ? (e ^ 0x80000000u) : ~e;
  return __uint_as_float(u);
}

// ------------------------------- utility -----------------------------------
__global__ void dct_fill_f32(float* p, float v, int n) {
  int i = blockIdx.x * 256 + threadIdx.x;
  if (i < n) p[i] = v;
}
__global__ void dct_fill_u32(unsigned* p, unsigned v, int n) {
  int i = blockIdx.x * 256 + threadIdx.x;
  if (i < n) p[i] = v;
}

// mean/var -> affine (s,t). If biasInStats==0, stats were accumulated pre-bias.
__global__ void dct_stats_finalize(const float* __restrict__ sum,
                                   const float* __restrict__ sq,
                                   const float* __restrict__ bias, int biasInStats,
                                   const float* __restrict__ g,
                                   const float* __restrict__ beta,
                                   float* __restrict__ s, float* __restrict__ t,
                                   int C, float invN) {
  int c = blockIdx.x * blockDim.x + threadIdx.x;
  if (c >= C) return;
  float S = sum[c] * invN, Q = sq[c] * invN;
  float mean, ex2;
  if (biasInStats) { mean = S; ex2 = Q; }
  else { float b = bias[c]; mean = S + b; ex2 = Q + 2.f * b * S + b * b; }
  float var = ex2 - mean * mean;
  float sc = g[c] * rsqrtf(var + EPSBN);
  s[c] = sc;
  t[c] = beta[c] - mean * sc;
}

// ------------------------- small VALU layers --------------------------------
// One thread per row; Cin<=32, Cout<=32. Stats (incl. bias) via LDS atomics.
// rotStart>=0: gather 6 cols from the 9-wide corners array with wraparound.
__global__ void dct_small_layer(const float* __restrict__ Xin, int Cin, int Cout,
                                int rotStart,
                                const float* __restrict__ inS, const float* __restrict__ inT,
                                const float* __restrict__ Wt, const float* __restrict__ bias,
                                float* __restrict__ Y,
                                float* __restrict__ gsum, float* __restrict__ gsq,
                                int N) {
  __shared__ float bs[32];
  __shared__ float bq[32];
  int tid = threadIdx.x;
  if (tid < 32) { bs[tid] = 0.f; bq[tid] = 0.f; }
  __syncthreads();
  int row = blockIdx.x * blockDim.x + tid;
  if (row < N) {
    float x[32];
    if (rotStart >= 0) {
      for (int j = 0; j < 6; ++j) {
        int c = rotStart + j; if (c >= 9) c -= 9;
        x[j] = Xin[(size_t)row * 9 + c];
      }
    } else {
      for (int k = 0; k < Cin; ++k) {
        float v = Xin[(size_t)row * Cin + k];
        if (inS) v = fmaxf(v * inS[k] + inT[k], 0.f);
        x[k] = v;
      }
    }
    for (int c = 0; c < Cout; ++c) {
      float a = bias[c];
      const float* wr = Wt + (size_t)c * Cin;
      for (int k = 0; k < Cin; ++k) a += x[k] * wr[k];
      Y[(size_t)row * Cout + c] = a;
      atomicAdd(&bs[c], a);
      atomicAdd(&bq[c], a * a);
    }
  }
  __syncthreads();
  if (tid < Cout) { atomicAdd(&gsum[tid], bs[tid]); atomicAdd(&gsq[tid], bq[tid]); }
}

// fea += relu(raw*s+t)/3   (rotate-branch averaging, 32-wide)
__global__ void dct_rot_accum(float* __restrict__ fea, const float* __restrict__ raw,
                              const float* __restrict__ s, const float* __restrict__ t,
                              size_t n) {
  size_t i = (size_t)blockIdx.x * 256 + threadIdx.x;
  if (i >= n) return;
  int c = (int)(i & 31);
  fea[i] += fmaxf(raw[i] * s[c] + t[c], 0.f) * (1.f / 3.f);
}

// ----------------------------- WMMA GEMM ------------------------------------
// Y[N,Cout] = act(X[N,Cin]) @ W[Cout,Cin]^T  (+bias stored, stats pre-bias)
// act = relu(x*inS+inT) when inS!=nullptr. Block tile 128 x (32*NJ), K-step 32.
// 8 wave32 waves in a 4x2 grid; each wave owns 32 rows x (16*NJ) cols.
#define BM 128
#define BK 32
#define LDX 40   // half-word stride: 80B/row -> conflict-free b128 frag reads
#define LDW 40

union FragU { unsigned u[8]; uint4 q[2]; v16bf v; };

template <int NJ>
__global__ __launch_bounds__(256)
void dct_wmma_gemm(const float* __restrict__ X, const float* __restrict__ Wt,
                   const float* __restrict__ bias,
                   const float* __restrict__ inS, const float* __restrict__ inT,
                   float* __restrict__ Y,
                   float* __restrict__ gsum, float* __restrict__ gsq,
                   int N, int Cin, int Cout) {
  constexpr int BNT = 32 * NJ;                 // block N-tile
  __shared__ __align__(16) unsigned short Xs[BM * LDX];
  __shared__ __align__(16) unsigned short Ws[BNT * LDW];
  int tid  = threadIdx.x;
  int lane = tid & 31, wid = tid >> 5;
  int wr = wid & 3, wc = wid >> 2;             // 4x2 wave grid
  int row0 = blockIdx.x * BM, col0 = blockIdx.y * BNT;
  int half = lane >> 4, l16 = lane & 15;
  const bool fullM = (row0 + BM <= N);

  v8f acc[2][NJ];
  for (int i = 0; i < 2; ++i)
    for (int j = 0; j < NJ; ++j)
      for (int e = 0; e < 8; ++e) acc[i][j][e] = 0.f;

  for (int k0 = 0; k0 < Cin; k0 += BK) {
    // ---- stage X tile: float4 loads, BN+ReLU, pack -> b64 LDS stores ----
    for (int slot = tid; slot < BM * (BK / 4); slot += 256) {
      int r = slot >> 3, c4 = (slot & 7) * 4;
      int gr = row0 + r, gc = k0 + c4;
      float4 v = make_float4(0.f, 0.f, 0.f, 0.f);
      if (fullM || gr < N) v = *(const float4*)(X + (size_t)gr * Cin + gc);
      if (inS) {
        float4 s4 = *(const float4*)(inS + gc);
        float4 t4 = *(const float4*)(inT + gc);
        v.x = fmaxf(v.x * s4.x + t4.x, 0.f);
        v.y = fmaxf(v.y * s4.y + t4.y, 0.f);
        v.z = fmaxf(v.z * s4.z + t4.z, 0.f);
        v.w = fmaxf(v.w * s4.w + t4.w, 0.f);
      }
      *(uint2*)(Xs + r * LDX + c4) = pack4bf(v.x, v.y, v.z, v.w);
    }
    // ---- stage W tile (rows of W = output columns) ----
    for (int slot = tid; slot < BNT * (BK / 4); slot += 256) {
      int r = slot >> 3, c4 = (slot & 7) * 4;
      float4 v = *(const float4*)(Wt + (size_t)(col0 + r) * Cin + k0 + c4);
      *(uint2*)(Ws + r * LDW + c4) = pack4bf(v.x, v.y, v.z, v.w);
    }
    __syncthreads();

    FragU fa[2], fb[NJ];
    // A 16x32 bf16: lane row = l16, K chunks [8h..8h+7] and [8h+16..8h+23]
    for (int i = 0; i < 2; ++i) {
      const unsigned short* base = Xs + (wr * 32 + 16 * i + l16) * LDX + 8 * half;
      fa[i].q[0] = *(const uint4*)base;
      fa[i].q[1] = *(const uint4*)(base + 16);
    }
    // B 32x16 bf16: lane col = l16, K chunks [8h..8h+7] and [8h+16..8h+23]
    for (int j = 0; j < NJ; ++j) {
      const unsigned short* base = Ws + (wc * 16 * NJ + 16 * j + l16) * LDW + 8 * half;
      fb[j].q[0] = *(const uint4*)base;
      fb[j].q[1] = *(const uint4*)(base + 16);
    }
    for (int i = 0; i < 2; ++i)
      for (int j = 0; j < NJ; ++j)
        acc[i][j] = __builtin_amdgcn_wmma_f32_16x16x32_bf16(
            false, fa[i].v, false, fb[j].v, (short)0, acc[i][j], false, false);
    __syncthreads();
  }

  // ---- epilogue: store y = acc + bias; reduce pre-bias column stats ----
  for (int i = 0; i < 2; ++i) {
    int rbase = row0 + wr * 32 + 16 * i + 8 * half;   // C/D: row = vgpr + 8*half
    for (int j = 0; j < NJ; ++j) {
      int col = col0 + wc * 16 * NJ + 16 * j + l16;   // C/D: col = lane%16
      float bcol = bias[col];
      float ps = 0.f, pq = 0.f;
      if (fullM) {
        float* yp = Y + (size_t)rbase * Cout + col;
        for (int e = 0; e < 8; ++e) {
          float val = acc[i][j][e];
          ps += val; pq += val * val;
          *yp = val + bcol;
          yp += Cout;
        }
      } else {
        for (int e = 0; e < 8; ++e) {
          float val = acc[i][j][e];
          ps += val; pq += val * val;
          int r = rbase + e;
          if (r < N) Y[(size_t)r * Cout + col] = val + bcol;
        }
      }
      float ps2 = __shfl_down(ps, 16, 32);
      float pq2 = __shfl_down(pq, 16, 32);
      if (half == 0) {
        atomicAdd(&gsum[col], ps + ps2);
        atomicAdd(&gsq[col],  pq + pq2);
      }
    }
  }
}

// --------------------------- concat + BN + ReLU -----------------------------
__global__ void dct_concat(const float* __restrict__ cen, const float* sc, const float* tc,
                           const float* __restrict__ edg, const float* se, const float* te,
                           const float* __restrict__ fus, const float* sf, const float* tf,
                           const float* __restrict__ fea, const float* sg, const float* tg,
                           float* __restrict__ X, int N) {
  size_t i = (size_t)blockIdx.x * 256 + threadIdx.x;
  if (i >= (size_t)N * 256) return;
  int col = (int)(i & 255);
  size_t row = i >> 8;
  float v;
  if (col < 32)       { int c = col;       v = fmaxf(cen[row * 32 + c] * sc[c] + tc[c], 0.f); }
  else if (col < 64)  { int c = col - 32;  v = fmaxf(edg[row * 32 + c] * se[c] + te[c], 0.f); }
  else if (col < 128) { int c = col - 64;  v = fmaxf(fus[row * 64 + c] * sf[c] + tf[c], 0.f); }
  else                { int c = col - 128; v = fmaxf(fea[row * 128 + c] * sg[c] + tg[c], 0.f); }
  X[i] = v;
}

// ------------------------- segment max (sorted idx) -------------------------
__global__ void dct_segmax(const float* __restrict__ Y,
                           const float* __restrict__ s, const float* __restrict__ t,
                           const int* __restrict__ idx, unsigned* __restrict__ out,
                           int N, int C) {
  __shared__ int sidx[64];
  int r0 = blockIdx.x * 64;
  int cnt = N - r0; if (cnt > 64) cnt = 64;
  if (threadIdx.x < 64 && threadIdx.x < cnt) sidx[threadIdx.x] = idx[r0 + threadIdx.x];
  __syncthreads();
  int c = blockIdx.y * blockDim.x + threadIdx.x;
  if (c >= C) return;
  float sc = s[c], tc = t[c];
  int cur = sidx[0];
  float m = -INFINITY;
  for (int r = 0; r < cnt; ++r) {
    int ci = sidx[r];
    if (ci != cur) { atomicMax(&out[(size_t)cur * C + c], encf(m)); cur = ci; m = -INFINITY; }
    m = fmaxf(m, Y[(size_t)(r0 + r) * C + c] * sc + tc);
  }
  atomicMax(&out[(size_t)cur * C + c], encf(m));
}

__global__ void dct_decode(const unsigned* __restrict__ in, float* __restrict__ out, size_t n) {
  size_t i = (size_t)blockIdx.x * 256 + threadIdx.x;
  if (i < n) out[i] = decf(in[i]);
}

// ----------------- final 256->230 linear + log_softmax ----------------------
__global__ __launch_bounds__(256)
void dct_head_final(const float* __restrict__ h2,
                    const float* __restrict__ s, const float* __restrict__ t,
                    const float* __restrict__ Wt, const float* __restrict__ bias,
                    float* __restrict__ out) {
  __shared__ float h[256];
  __shared__ float red[256];
  int row = blockIdx.x, tid = threadIdx.x;
  h[tid] = fmaxf(h2[(size_t)row * 256 + tid] * s[tid] + t[tid], 0.f);
  __syncthreads();
  float acc = 0.f;
  if (tid < 230) {
    const float* wr = Wt + (size_t)tid * 256;
    for (int k = 0; k < 256; ++k) acc += h[k] * wr[k];
    acc += bias[tid];
  }
  red[tid] = (tid < 230) ? acc : -INFINITY;
  __syncthreads();
  for (int o = 128; o > 0; o >>= 1) { if (tid < o) red[tid] = fmaxf(red[tid], red[tid + o]); __syncthreads(); }
  float mx = red[0];
  __syncthreads();
  red[tid] = (tid < 230) ? expf(acc - mx) : 0.f;
  __syncthreads();
  for (int o = 128; o > 0; o >>= 1) { if (tid < o) red[tid] += red[tid + o]; __syncthreads(); }
  float lse = logf(red[0]);
  if (tid < 230) out[(size_t)row * 230 + tid] = acc - mx - lse;
}

// ---------------------------------------------------------------------------
extern "C" void kernel_launch(void* const* d_in, const int* in_sizes, int n_in,
                              void* d_out, int out_size, void* d_ws, size_t ws_size,
                              hipStream_t stream) {
  (void)n_in; (void)ws_size;
#define FP(i) ((const float*)d_in[(i)])
  // params flattened in setup_inputs() dict insertion order:
  // rotate[2], fusion[2], centers[2], edges[2], feats[2], trunk[2], head[3]
  // each layer: W,b,g,beta (final head layer: W,b only)
  const float *rW0=FP(0),  *rb0=FP(1),  *rg0=FP(2),  *rbe0=FP(3);
  const float *rW1=FP(4),  *rb1=FP(5),  *rg1=FP(6),  *rbe1=FP(7);
  const float *fW0=FP(8),  *fb0=FP(9),  *fg0=FP(10), *fbe0=FP(11);
  const float *fW1=FP(12), *fb1=FP(13), *fg1=FP(14), *fbe1=FP(15);
  const float *cW0=FP(16), *cb0=FP(17), *cg0=FP(18), *cbe0=FP(19);
  const float *cW1=FP(20), *cb1=FP(21), *cg1=FP(22), *cbe1=FP(23);
  const float *eW0=FP(24), *eb0=FP(25), *eg0=FP(26), *ebe0=FP(27);
  const float *eW1=FP(28), *eb1=FP(29), *eg1=FP(30), *ebe1=FP(31);
  const float *ftW0=FP(32),*ftb0=FP(33),*ftg0=FP(34),*ftbe0=FP(35);
  const float *ftW1=FP(36),*ftb1=FP(37),*ftg1=FP(38),*ftbe1=FP(39);
  const float *tW0=FP(40), *tb0=FP(41), *tg0=FP(42), *tbe0=FP(43);
  const float *tW1=FP(44), *tb1=FP(45), *tg1=FP(46), *tbe1=FP(47);
  const float *hW0=FP(48), *hb0=FP(49), *hg0=FP(50), *hbe0=FP(51);
  const float *hW1=FP(52), *hb1=FP(53), *hg1=FP(54), *hbe1=FP(55);
  const float *hW2=FP(56), *hb2=FP(57);
  const float *in_centers  = FP(58);
  const float *in_features = FP(59);
  const float *in_corners  = FP(60);
  const float *in_edges    = FP(61);
  const int   *cry         = (const int*)d_in[62];

  const int N    = in_sizes[58] / 3;      // 250000 faces
  const int numC = out_size / 230;        // 2048 crystals
  const float invN = 1.f / (float)N;
  const float invC = 1.f / (float)numC;

  // ---- workspace layout (floats) ----
  float* ws = (float*)d_ws;
  size_t o = 0;
  auto take = [&](size_t n) { float* p = ws + o; o += n; return p; };
  struct St { float *sum, *sq, *s, *t; };
  auto stat = [&](int C) { St st; st.sum = take(C); st.sq = take(C); st.s = take(C); st.t = take(C); return st; };
  St sCen1 = stat(16),  sCen2 = stat(32), sEdg1 = stat(16), sEdg2 = stat(32);
  St sRot1 = stat(32),  sRot2 = stat(32);
  St sFus1 = stat(64),  sFus2 = stat(64);
  St sFe1  = stat(64),  sFe2  = stat(128);
  St sTr1  = stat(512), sTr2  = stat(1024);
  St sHd1  = stat(512), sHd2  = stat(256);
  float* cen1  = take((size_t)N * 16);
  float* cen2  = take((size_t)N * 32);
  float* edg1  = take((size_t)N * 16);
  float* edg2  = take((size_t)N * 32);
  float* rotT1 = take((size_t)N * 32);
  float* rotT2 = take((size_t)N * 32);
  float* feaA  = take((size_t)N * 32);
  float* fus1  = take((size_t)N * 64);
  float* fus2  = take((size_t)N * 64);
  float* fe1   = take((size_t)N * 64);
  float* fe2   = take((size_t)N * 128);
  float* X256  = take((size_t)N * 256);
  float* trk1  = take((size_t)N * 512);
  float* trk2  = take((size_t)N * 1024);
  unsigned* pooled = (unsigned*)take((size_t)numC * 1024);
  float* Pm = take((size_t)numC * 1024);
  float* h1 = take((size_t)numC * 512);
  float* h2 = take((size_t)numC * 256);

  auto zeroStats = [&](St& st, int C) {
    dct_fill_f32<<<(2 * C + 255) / 256, 256, 0, stream>>>(st.sum, 0.f, 2 * C); // sum+sq contiguous
  };
  auto finalize = [&](St& st, const float* bias, int biasIn, const float* g, const float* beta,
                      int C, float inv) {
    dct_stats_finalize<<<(C + 63) / 64, 64, 0, stream>>>(st.sum, st.sq, bias, biasIn, g, beta,
                                                         st.s, st.t, C, inv);
  };
  auto smallL = [&](const float* X, int Cin, int Cout, int rotStart,
                    const float* iS, const float* iT,
                    const float* W, const float* b, float* Y, St& st) {
    zeroStats(st, Cout);
    dct_small_layer<<<(N + 255) / 256, 256, 0, stream>>>(X, Cin, Cout, rotStart, iS, iT, W, b, Y,
                                                         st.sum, st.sq, N);
  };
  auto wmmaL = [&](const float* X, int rows, int Cin, int Cout,
                   const float* iS, const float* iT,
                   const float* W, const float* b, float* Y, St& st) {
    zeroStats(st, Cout);
    if (Cout % 128 == 0) {
      dim3 g((rows + BM - 1) / BM, Cout / 128);
      dct_wmma_gemm<4><<<g, 256, 0, stream>>>(X, W, b, iS, iT, Y, st.sum, st.sq, rows, Cin, Cout);
    } else {
      dim3 g((rows + BM - 1) / BM, Cout / 64);
      dct_wmma_gemm<2><<<g, 256, 0, stream>>>(X, W, b, iS, iT, Y, st.sum, st.sq, rows, Cin, Cout);
    }
  };

  // ---- centers branch: 3->16->32 ----
  smallL(in_centers, 3, 16, -1, nullptr, nullptr, cW0, cb0, cen1, sCen1);
  finalize(sCen1, nullptr, 1, cg0, cbe0, 16, invN);
  smallL(cen1, 16, 32, -1, sCen1.s, sCen1.t, cW1, cb1, cen2, sCen2);
  finalize(sCen2, nullptr, 1, cg1, cbe1, 32, invN);

  // ---- edges branch: 3->16->32 ----
  smallL(in_edges, 3, 16, -1, nullptr, nullptr, eW0, eb0, edg1, sEdg1);
  finalize(sEdg1, nullptr, 1, eg0, ebe0, 16, invN);
  smallL(edg1, 16, 32, -1, sEdg1.s, sEdg1.t, eW1, eb1, edg2, sEdg2);
  finalize(sEdg2, nullptr, 1, eg1, ebe1, 32, invN);

  // ---- rotate branch: 3 slices through shared 6->32->32, averaged ----
  dct_fill_f32<<<((int)((size_t)N * 32) + 255) / 256, 256, 0, stream>>>(feaA, 0.f, N * 32);
  for (int sl = 0; sl < 3; ++sl) {
    smallL(in_corners, 6, 32, 3 * sl, nullptr, nullptr, rW0, rb0, rotT1, sRot1);
    finalize(sRot1, nullptr, 1, rg0, rbe0, 32, invN);
    smallL(rotT1, 32, 32, -1, sRot1.s, sRot1.t, rW1, rb1, rotT2, sRot2);
    finalize(sRot2, nullptr, 1, rg1, rbe1, 32, invN);
    size_t n32 = (size_t)N * 32;
    dct_rot_accum<<<(unsigned)((n32 + 255) / 256), 256, 0, stream>>>(feaA, rotT2, sRot2.s, sRot2.t, n32);
  }

  // ---- fusion: 32->64->64 (WMMA) ----
  wmmaL(feaA, N, 32, 64, nullptr, nullptr, fW0, fb0, fus1, sFus1);
  finalize(sFus1, fb0, 0, fg0, fbe0, 64, invN);
  wmmaL(fus1, N, 64, 64, sFus1.s, sFus1.t, fW1, fb1, fus2, sFus2);
  finalize(sFus2, fb1, 0, fg1, fbe1, 64, invN);

  // ---- feats: 64->64->128 (WMMA) ----
  wmmaL(in_features, N, 64, 64, nullptr, nullptr, ftW0, ftb0, fe1, sFe1);
  finalize(sFe1, ftb0, 0, ftg0, ftbe0, 64, invN);
  wmmaL(fe1, N, 64, 128, sFe1.s, sFe1.t, ftW1, ftb1, fe2, sFe2);
  finalize(sFe2, ftb1, 0, ftg1, ftbe1, 128, invN);

  // ---- concat [c|e|fr|f] with BN+ReLU -> X256 ----
  {
    size_t tot = (size_t)N * 256;
    dct_concat<<<(unsigned)((tot + 255) / 256), 256, 0, stream>>>(
        cen2, sCen2.s, sCen2.t, edg2, sEdg2.s, sEdg2.t,
        fus2, sFus2.s, sFus2.t, fe2, sFe2.s, sFe2.t, X256, N);
  }

  // ---- trunk: 256->512->1024 (WMMA) ----
  wmmaL(X256, N, 256, 512, nullptr, nullptr, tW0, tb0, trk1, sTr1);
  finalize(sTr1, tb0, 0, tg0, tbe0, 512, invN);
  wmmaL(trk1, N, 512, 1024, sTr1.s, sTr1.t, tW1, tb1, trk2, sTr2);
  finalize(sTr2, tb1, 0, tg1, tbe1, 1024, invN);

  // ---- per-crystal segment max (BN applied, no ReLU) ----
  dct_fill_u32<<<(numC * 1024 + 255) / 256, 256, 0, stream>>>(pooled, ENC_NEGINF, numC * 1024);
  {
    dim3 g((N + 63) / 64, 1024 / 256);
    dct_segmax<<<g, 256, 0, stream>>>(trk2, sTr2.s, sTr2.t, cry, pooled, N, 1024);
  }
  {
    size_t n = (size_t)numC * 1024;
    dct_decode<<<(unsigned)((n + 255) / 256), 256, 0, stream>>>(pooled, Pm, n);
  }

  // ---- head: 1024->512->256 (WMMA), then 256->230 + log_softmax ----
  wmmaL(Pm, numC, 1024, 512, nullptr, nullptr, hW0, hb0, h1, sHd1);
  finalize(sHd1, hb0, 0, hg0, hbe0, 512, invC);
  wmmaL(h1, numC, 512, 256, sHd1.s, sHd1.t, hW1, hb1, h2, sHd2);
  finalize(sHd2, hb1, 0, hg1, hbe1, 256, invC);

  dct_head_final<<<numC, 256, 0, stream>>>(h2, sHd2.s, sHd2.t, hW2, hb2, (float*)d_out);
}
